// DataEmbedding_68186900791519
// MI455X (gfx1250) — matrix operand order repaired and checked by reading, not compiled, not run on hardware
//
#include <hip/hip_runtime.h>
#include <hip/hip_bf16.h>

// ---------------- problem constants ----------------
#define B_SZ   64
#define L_SZ   3168
#define C_IN   64
#define D_MODEL 512
#define KWIN   16
#define W_SZ   198            // L / K
#define ROWS   (B_SZ * W_SZ)  // 12672
#define M_TILE 64
#define ROW_TILES (ROWS / M_TILE) // 198
#define KRED   (C_IN * KWIN)  // 1024
#define A_STRIDE 1040         // padded row stride (bf16 elems)

typedef __attribute__((ext_vector_type(16))) __bf16 v16bf;
typedef __attribute__((ext_vector_type(8)))  float  v8f;

union Frag16 {
    v16bf v;
    uint4 q[2];
};

__device__ __forceinline__ unsigned short f2bf(float f) {
    unsigned int u = __float_as_uint(f);
    unsigned int r = u + 0x7FFFu + ((u >> 16) & 1u);   // round to nearest even
    return (unsigned short)(r >> 16);
}

// ---------------- K0: w_def f32 -> bf16 (WMMA-ready) ----------------
__global__ void k_convert_wdef(const float* __restrict__ w, unsigned short* __restrict__ o) {
    int i = blockIdx.x * 1024 + threadIdx.x * 4;     // 512 blocks * 256 thr * 4 = 524288
    float4 v = *(const float4*)(w + i);
    o[i + 0] = f2bf(v.x);
    o[i + 1] = f2bf(v.y);
    o[i + 2] = f2bf(v.z);
    o[i + 3] = f2bf(v.w);
}

// ---------------- K1: offset GEMM  off[b,o,w] ----------------
// grid = 64 batches * 6 segments, 33 windows each; w_off (128KB) staged in dynamic LDS.
__global__ void k_offsets(const float* __restrict__ x,
                          const float* __restrict__ w_off,
                          const float* __restrict__ b_off,
                          float* __restrict__ off_ws) {
    extern __shared__ float smem[];
    float* s_w = smem;               // 32*1024 floats (128KB)
    float* s_x = smem + 32 * 1024;   // 1024 floats  (4KB)

    const int tid   = threadIdx.x;
    const int b     = blockIdx.x / 6;
    const int wbase = (blockIdx.x % 6) * 33;

    // cooperative load of w_off into LDS (8192 float4)
    const float4* wsrc = (const float4*)w_off;
    float4* wdst = (float4*)s_w;
    for (int i = tid; i < 8192; i += 256) wdst[i] = wsrc[i];

    const int o = tid >> 3;   // 0..31 output channel
    const int p = tid & 7;    // reduce split

    for (int wi = 0; wi < 33; ++wi) {
        const int w = wbase + wi;
        __syncthreads();  // w_off visible; previous window's consumers done
        // xr window: 1024 contiguous floats
        const float4* xs = (const float4*)(x + ((size_t)b * L_SZ + (size_t)w * KWIN) * C_IN);
        ((float4*)s_x)[tid] = xs[tid];
        __syncthreads();

        float acc = 0.f;
#pragma unroll
        for (int cc = 0; cc < 8; ++cc) {
            const int c = p * 8 + cc;
#pragma unroll
            for (int k = 0; k < 16; ++k)
                acc += s_x[k * 64 + c] * s_w[o * 1024 + c * 16 + k];
        }
        acc += __shfl_xor(acc, 1, 32);
        acc += __shfl_xor(acc, 2, 32);
        acc += __shfl_xor(acc, 4, 32);
        if (p == 0)
            off_ws[((size_t)b * 32 + o) * W_SZ + w] = acc + b_off[o];
    }
}

// ---------------- K2: fused gather + blend + WMMA GEMM + epilogue ----------------
// 198 blocks (64 rows each), 256 threads (8 waves), each wave owns 64 output cols
// and 4 M-subtiles -> each B fragment feeds 4 WMMAs (4x less L2 B traffic).
__global__ void __launch_bounds__(256)
k_main(const float* __restrict__ x,
       const float* __restrict__ off_ws,
       const unsigned short* __restrict__ wdefb,
       const float* __restrict__ b_def,
       float* __restrict__ out) {
    // A-tile: 64 rows x 1024 (kk = c*16 + k), bf16, padded row stride 1040 (130KB dynamic LDS)
    extern __shared__ unsigned short sA[];

    const int tid = threadIdx.x;

    // ---------- phase 1: build A tile (4 (row,k) pairs per thread) ----------
#pragma unroll
    for (int it = 0; it < 4; ++it) {
        const int p   = tid + 256 * it;
        const int row = p >> 4;          // 0..63
        const int k   = p & 15;
        const int gr  = blockIdx.x * M_TILE + row;
        const int b   = gr / W_SZ;
        const int w   = gr % W_SZ;

        const float dy = off_ws[((size_t)b * 32 + 2 * k)     * W_SZ + w];
        const float dx = off_ws[((size_t)b * 32 + 2 * k + 1) * W_SZ + w];

        const float px   = (float)(w * KWIN + k) + dx;
        const float x0f  = floorf(px);
        const float frac = px - x0f;
        const int   x0i  = (int)x0f;
        const float wy   = fmaxf(0.f, 1.f - fabsf(dy));
        const float a0   = wy * (1.f - frac);
        const float a1   = wy * frac;

        const bool v0ok = (x0i >= 0) && (x0i < L_SZ);
        const bool v1ok = (x0i + 1 >= 0) && (x0i + 1 < L_SZ);
        const int  c0   = min(max(x0i, 0), L_SZ - 1);
        const int  c1   = min(max(x0i + 1, 0), L_SZ - 1);
        const float4* p0 = (const float4*)(x + ((size_t)b * L_SZ + c0) * C_IN);
        const float4* p1 = (const float4*)(x + ((size_t)b * L_SZ + c1) * C_IN);
        const float4 z4 = make_float4(0.f, 0.f, 0.f, 0.f);

#pragma unroll
        for (int c = 0; c < 64; c += 4) {
            float4 u0 = v0ok ? p0[c >> 2] : z4;
            float4 u1 = v1ok ? p1[c >> 2] : z4;
            unsigned short* dst = &sA[row * A_STRIDE + c * 16 + k];
            dst[0 * 16] = f2bf(a0 * u0.x + a1 * u1.x);
            dst[1 * 16] = f2bf(a0 * u0.y + a1 * u1.y);
            dst[2 * 16] = f2bf(a0 * u0.z + a1 * u1.z);
            dst[3 * 16] = f2bf(a0 * u0.w + a1 * u1.w);
        }
    }
    __syncthreads();

    // ---------- phase 2: 64x64 tile per wave via v_wmma_f32_16x16x32_bf16 ----------
    const int lane = tid & 31;
    const int wv   = tid >> 5;
    const int d0   = wv * 64;
    const int r    = lane & 15;   // A: M row / B,D: N col
    const int h    = lane >> 4;   // half-wave selector

    v8f acc[4][4] = {};           // [m-subtile][n-tile]

    for (int ks = 0; ks < 32; ++ks) {
        const int kb = ks * 32;

        // A fragments for the 4 M-subtiles
        // (16-bit A 16x32 lane layout: elems 0..7 -> K = h*8+e, 8..15 -> K = 16+h*8+e)
        Frag16 af[4];
#pragma unroll
        for (int ms = 0; ms < 4; ++ms) {
            const int arow = (ms * 16 + r) * A_STRIDE + kb + h * 8;
            af[ms].q[0] = *(const uint4*)&sA[arow];
            af[ms].q[1] = *(const uint4*)&sA[arow + 16];
        }

#pragma unroll
        for (int t = 0; t < 4; ++t) {
            // B fragment (32x16): lane n = r holds column d0+t*16+r, elems 0..7 -> K = kb + h*8 + e, 8..15 -> K = kb + 16 + h*8 + e
            const unsigned short* bp =
                wdefb + (size_t)(d0 + t * 16 + r) * KRED + kb + h * 8;
            __builtin_prefetch(bp + 256, 0, 1);   // global_prefetch for the w_def stream
            Frag16 bf;
            bf.q[0] = *(const uint4*)bp;
            bf.q[1] = *(const uint4*)(bp + 16);

#pragma unroll
            for (int ms = 0; ms < 4; ++ms)
                acc[ms][t] = __builtin_amdgcn_wmma_f32_16x16x32_bf16(
                    false, af[ms].v, false, bf.v, (short)0, acc[ms][t], false, false);
        }
    }

    // ---------- epilogue: bias + positional encoding, store ----------
    const float kNegLog = -0.017988946f;  // -ln(10000)/512
#pragma unroll
    for (int t = 0; t < 4; ++t) {
        const int d = d0 + t * 16 + r;
        const float bias = b_def[d];
        const float freq = __expf((float)(d & ~1) * kNegLog);
#pragma unroll
        for (int ms = 0; ms < 4; ++ms) {
#pragma unroll
            for (int j = 0; j < 8; ++j) {
                const int m  = ms * 16 + j + h * 8;   // C/D layout: lanes 16..31 hold M = j+8
                const int gr = blockIdx.x * M_TILE + m;
                const int b  = gr / W_SZ;
                const int w  = gr % W_SZ;
                const int s  = w + 1;                 // token position (row 0 is cls)
                float sv, cv;
                __sincosf((float)s * freq, &sv, &cv);
                const float pe = (d & 1) ? cv : sv;
                out[((size_t)b * (W_SZ + 1) + s) * D_MODEL + d] = acc[ms][t][j] + bias + pe;
            }
        }
    }
}

// ---------------- K3: cls-token row + pe[0] ----------------
__global__ void k_cls(const float* __restrict__ cls, float* __restrict__ out) {
    const int i = blockIdx.x * 256 + threadIdx.x;   // 0 .. 32767
    const int b = i >> 9;
    const int d = i & 511;
    const float pe0 = (d & 1) ? 1.0f : 0.0f;        // cos(0)=1 (odd), sin(0)=0 (even)
    out[(size_t)b * (W_SZ + 1) * D_MODEL + d] = cls[d] + pe0;
}

// ---------------- launcher ----------------
extern "C" void kernel_launch(void* const* d_in, const int* in_sizes, int n_in,
                              void* d_out, int out_size, void* d_ws, size_t ws_size,
                              hipStream_t stream) {
    const float* x     = (const float*)d_in[0];   // (64,3168,64)
    const float* w_off = (const float*)d_in[1];   // (32,64,16)
    const float* b_off = (const float*)d_in[2];   // (32,)
    const float* w_def = (const float*)d_in[3];   // (512,64,16)
    const float* b_def = (const float*)d_in[4];   // (512,)
    const float* cls   = (const float*)d_in[5];   // (1,1,512)
    float* out = (float*)d_out;                   // (64,199,512)

    // workspace: off[b,32,w] floats, then w_def in bf16
    float* off_ws = (float*)d_ws;
    const size_t off_bytes = (size_t)B_SZ * 32 * W_SZ * sizeof(float);   // 1,622,016 (256-aligned)
    unsigned short* wdefb = (unsigned short*)((char*)d_ws + off_bytes);

    k_convert_wdef<<<512, 256, 0, stream>>>(w_def, wdefb);

    const size_t lds1 = (32 * 1024 + 1024) * sizeof(float);  // 132KB dynamic LDS
    k_offsets<<<B_SZ * 6, 256, lds1, stream>>>(x, w_off, b_off, off_ws);

    const size_t lds2 = (size_t)M_TILE * A_STRIDE * sizeof(unsigned short); // 133,120B
    k_main<<<ROW_TILES, 256, lds2, stream>>>(x, off_ws, wdefb, b_def, out);

    k_cls<<<(B_SZ * D_MODEL) / 256, 256, 0, stream>>>(cls, out);
}
